// RNN_14413910245604
// MI455X (gfx1250) — matrix operand order repaired and checked by reading, not compiled, not run on hardware
//
#include <hip/hip_runtime.h>
#include <hip/hip_bf16.h>
#include <math.h>

// ---------------------------------------------------------------------------
// LSTM (T=2048, B=16, IN=72, H=256, OUT=90, L=2) for gfx1250 (MI455X).
// bf16 WMMA everywhere. Persistent single-WGP scan: 16 waves; Whh split
// half register-resident (128 VGPRs/lane) + half LDS-resident (256 KB),
// staged via global_load_async_to_lds_b128. h ping-pongs through LDS.
// ---------------------------------------------------------------------------

typedef __attribute__((ext_vector_type(16))) __bf16 v16bf;
typedef __attribute__((ext_vector_type(8)))  float  v8f;

#define T_DIM 2048
#define B_DIM 16
#define IN_DIM 72
#define KP_DIM 96      // IN padded to multiple of 32
#define H_DIM 256
#define G4_DIM 1024    // 4*H
#define OUT_DIM 90
#define NP_DIM 96      // OUT padded to multiple of 16
#define M_DIM (T_DIM*B_DIM)

static __device__ __forceinline__ float sigf(float x) {
    return 1.0f / (1.0f + __expf(-x));
}

static __device__ __forceinline__ uint32_t lds_addr32(const void* p) {
    // flat shared address -> 32-bit LDS byte offset (aperture rule: addr[31:0])
    return (uint32_t)(uintptr_t)p;
}

// ------------------------------- prep kernels ------------------------------

__global__ void k_pad_x(const float* __restrict__ x, __bf16* __restrict__ xb) {
    int idx = blockIdx.x * blockDim.x + threadIdx.x;
    int total = M_DIM * KP_DIM;
    if (idx >= total) return;
    int row = idx / KP_DIM, col = idx % KP_DIM;
    float v = (col < IN_DIM) ? x[(size_t)row * IN_DIM + col] : 0.0f;
    xb[idx] = (__bf16)v;
}

__global__ void k_pad2d(const float* __restrict__ src, __bf16* __restrict__ dst,
                        int csrc, int cdst, int rvalid, int total) {
    int idx = blockIdx.x * blockDim.x + threadIdx.x;
    if (idx >= total) return;
    int r = idx / cdst, c = idx % cdst;
    float v = (r < rvalid && c < csrc) ? src[(size_t)r * csrc + c] : 0.0f;
    dst[idx] = (__bf16)v;
}

__global__ void k_conv(const float* __restrict__ s, __bf16* __restrict__ d, int n) {
    int idx = blockIdx.x * blockDim.x + threadIdx.x;
    if (idx < n) d[idx] = (__bf16)s[idx];
}

__global__ void k_addvec(const float* __restrict__ a, const float* __restrict__ b,
                         float* __restrict__ d, int n) {
    int idx = blockIdx.x * blockDim.x + threadIdx.x;
    if (idx < n) d[idx] = a[idx] + b[idx];
}

__global__ void k_padbias(const float* __restrict__ s, float* __restrict__ d,
                          int nvalid, int n) {
    int idx = blockIdx.x * blockDim.x + threadIdx.x;
    if (idx < n) d[idx] = (idx < nvalid) ? s[idx] : 0.0f;
}

// ------------------------------- WMMA GEMM ---------------------------------
// C[M,N] = A[M,K] @ W[N,K]^T + bias[N]   (A bf16 row-major, W bf16 row-major)
// KCN = K/32 (compile time).  MODE:
//   0: bf16 out, row-major ldo (+RELU)
//   1: bf16 out, transposed-within-16-rows: out[row0*N + col*16 + m]
//   2: f32 out, ldo stride, cols < colsValid, masked by (t < lengths[b])
template <int KCN, int MODE, int RELU>
__global__ void __launch_bounds__(256)
wmma_gemm(const __bf16* __restrict__ A, const __bf16* __restrict__ W,
          const float* __restrict__ bias, void* __restrict__ outp,
          const int* __restrict__ lengths, int N, int ldo, int colsValid) {
    constexpr int K = KCN * 32;
    __shared__ __bf16 atile[16 * K];
    const int tid = threadIdx.x;
    const int row0 = blockIdx.x * 16;

    // stage the contiguous 16-row A tile into LDS
    {
        const uint4* src = (const uint4*)(A + (size_t)row0 * K);
        uint4* dst = (uint4*)atile;
        constexpr int chunks = 2 * K;  // (16*K)/8 bf16 per uint4
#pragma unroll
        for (int i = 0; i < (chunks + 255) / 256; ++i) {
            int idx = tid + i * 256;
            if (chunks % 256 == 0 || idx < chunks) dst[idx] = src[idx];
        }
    }
    __syncthreads();

    const int wv = tid >> 5, lane = tid & 31;
    const int n = lane & 15, hf = lane >> 4;

    // A fragments (16x32 bf16 layout: per lane m=lane&15, two 8-elem runs at
    // kc*32 + 16*hf and +8)
    v16bf af[KCN];
#pragma unroll
    for (int kc = 0; kc < KCN; kc++) {
        union { v16bf v; uint4 q[2]; } u;
        const __bf16* ap = atile + n * K + kc * 32 + hf * 16;
        u.q[0] = *(const uint4*)ap;
        u.q[1] = *(const uint4*)(ap + 8);
        af[kc] = u.v;
    }

#pragma unroll
    for (int tt = 0; tt < 4; tt++) {
        const int n0 = (blockIdx.y * 8 + wv) * 64 + tt * 16;
        if (n0 >= N) continue;
        const __bf16* wp = W + (size_t)(n0 + n) * K + hf * 16;
        v8f acc = {0.f, 0.f, 0.f, 0.f, 0.f, 0.f, 0.f, 0.f};
#pragma unroll
        for (int kc = 0; kc < KCN; kc++) {
            // B fragment (32x16 bf16): lane n=lane&15, khalf=lane>>4,
            // contiguous 16 elements of row (n0+n)
            v16bf bfr = *(const v16bf*)(wp + kc * 32);
            acc = __builtin_amdgcn_wmma_f32_16x16x32_bf16(
                false, af[kc], false, bfr, (short)0, acc, false, false);
        }
        const float bv = bias[n0 + n];
        if (MODE == 0) {
            __bf16* ob = (__bf16*)outp;
#pragma unroll
            for (int v = 0; v < 8; v++) {
                float val = acc[v] + bv;
                if (RELU) val = fmaxf(val, 0.0f);
                ob[(size_t)(row0 + v + 8 * hf) * ldo + n0 + n] = (__bf16)val;
            }
        } else if (MODE == 1) {
            __bf16* ob = (__bf16*)outp;
#pragma unroll
            for (int v = 0; v < 8; v++) {
                float val = acc[v] + bv;
                ob[(size_t)row0 * N + (size_t)(n0 + n) * 16 + v + 8 * hf] = (__bf16)val;
            }
        } else {
            float* of = (float*)outp;
            const int col = n0 + n;
            if (col < colsValid) {
                const int time = row0 >> 4;  // B==16: one timestep per block
#pragma unroll
                for (int v = 0; v < 8; v++) {
                    const int b = v + 8 * hf;
                    float val = acc[v] + bv;
                    if (!(time < lengths[b])) val = 0.0f;
                    of[(size_t)(row0 + b) * ldo + col] = val;
                }
            }
        }
    }
}

// --------------------------- persistent LSTM scan --------------------------
// gi:  [T][1024 cols][16 batch] bf16 (bih+bhh folded in)
// Whhb:[1024][256] bf16 row-major
// ys:  [T*16][256] bf16 (masked layer output)
// One block, 512 threads = 16 waves; wave w owns hidden units [w*16, w*16+16).
// Whh fragments: k-chunks 0..3 in registers (128 VGPRs/lane), k-chunks 4..7
// in LDS (256 KB), staged once via global_load_async_to_lds_b128 (ASYNCcnt).
// LDS weight block per fragment: 1 KB = two 512B halves, lane-stride 16B.
__global__ void __launch_bounds__(512)
lstm_recurrent(const __bf16* __restrict__ gi, const __bf16* __restrict__ Whhb,
               const int* __restrict__ lengths, __bf16* __restrict__ ys) {
    __shared__ __bf16 hbuf[2][16 * H_DIM];                          // 16 KB
    __shared__ __attribute__((aligned(16))) char wlds[256 * 1024];  // 256 KB
    const int tid = threadIdx.x;
    const int wv = tid >> 5, lane = tid & 31;
    const int n = lane & 15, hf = lane >> 4;
    const int u0 = wv * 16;

    // zero h(t=0)
    for (int i = tid; i < 2 * 16 * H_DIM; i += 512) ((__bf16*)hbuf)[i] = (__bf16)0.0f;

    // async-stage this wave's LDS-resident weight fragments (gates x kc 4..7):
    // permuted scatter is free since each lane supplies its own LDS address
#pragma unroll
    for (int g = 0; g < 4; ++g) {
#pragma unroll
        for (int kc4 = 0; kc4 < 4; ++kc4) {
            const size_t gbyte =
                ((size_t)(g * 256 + u0 + n) * H_DIM + (kc4 + 4) * 32 + hf * 16) * 2;
            const int fragbase = (wv * 16 + g * 4 + kc4) * 1024;
            uint64_t ga = (uint64_t)(uintptr_t)((const char*)Whhb + gbyte);
            uint32_t la0 = lds_addr32(wlds + fragbase + lane * 16);
            uint32_t la1 = lds_addr32(wlds + fragbase + 512 + lane * 16);
            asm volatile("global_load_async_to_lds_b128 %0, %1, off"
                         : : "v"(la0), "v"(ga) : "memory");
            asm volatile("global_load_async_to_lds_b128 %0, %1, off"
                         : : "v"(la1), "v"(ga + 16) : "memory");
        }
    }

    // register-resident Whh fragments: 4 gates x k-chunks 0..3
    v16bf wf[4][4];
#pragma unroll
    for (int g = 0; g < 4; ++g)
#pragma unroll
        for (int kc = 0; kc < 4; ++kc)
            wf[g][kc] = *(const v16bf*)(Whhb + (size_t)(g * 256 + u0 + n) * H_DIM +
                                        kc * 32 + hf * 16);

    int lenv[8];
    float cacc[8], hreg[8];
#pragma unroll
    for (int v = 0; v < 8; ++v) {
        lenv[v] = lengths[v + 8 * hf];
        cacc[v] = 0.0f;
        hreg[v] = 0.0f;
    }

    asm volatile("s_wait_asynccnt 0" ::: "memory");
    __syncthreads();

    for (int t = 0; t < T_DIM; ++t) {
        const int cur = t & 1, nxt = cur ^ 1;

        // init accumulators from gi in global (contiguous 16B per gate, L2-hot)
        v8f acc[4];
#pragma unroll
        for (int g = 0; g < 4; ++g) {
            union { uint4 q; __bf16 h[8]; } gq;
            gq.q = *(const uint4*)(gi + (size_t)t * 16384 +
                                   (size_t)(g * 256 + u0 + n) * 16 + hf * 8);
            v8f a;
#pragma unroll
            for (int v = 0; v < 8; ++v) a[v] = (float)gq.h[v];
            acc[g] = a;
        }
        if (t + 1 < T_DIM) {
#pragma unroll
            for (int g = 0; g < 4; ++g)
                __builtin_prefetch(gi + (size_t)(t + 1) * 16384 +
                                       (size_t)(g * 256 + u0 + n) * 16 + hf * 8,
                                   0, 3);
        }

        // k-chunks 0..3: register weights
#pragma unroll
        for (int kc = 0; kc < 4; ++kc) {
            union { v16bf v; uint4 q[2]; } au;
            const __bf16* hp = &hbuf[cur][n * H_DIM + kc * 32 + hf * 16];
            au.q[0] = *(const uint4*)hp;
            au.q[1] = *(const uint4*)(hp + 8);
#pragma unroll
            for (int g = 0; g < 4; ++g)
                acc[g] = __builtin_amdgcn_wmma_f32_16x16x32_bf16(
                    false, au.v, false, wf[g][kc], (short)0, acc[g], false, false);
        }

        // k-chunks 4..7: LDS weights
#pragma unroll
        for (int kc4 = 0; kc4 < 4; ++kc4) {
            union { v16bf v; uint4 q[2]; } au;
            const __bf16* hp = &hbuf[cur][n * H_DIM + (kc4 + 4) * 32 + hf * 16];
            au.q[0] = *(const uint4*)hp;
            au.q[1] = *(const uint4*)(hp + 8);
#pragma unroll
            for (int g = 0; g < 4; ++g) {
                const int fragbase = (wv * 16 + g * 4 + kc4) * 1024;
                union { v16bf v; uint4 q[2]; } wu;
                wu.q[0] = *(const uint4*)(wlds + fragbase + lane * 16);
                wu.q[1] = *(const uint4*)(wlds + fragbase + 512 + lane * 16);
                acc[g] = __builtin_amdgcn_wmma_f32_16x16x32_bf16(
                    false, au.v, false, wu.v, (short)0, acc[g], false, false);
            }
        }

        // gates + state update (C layout: row m = v + 8*hf, col = u0+n)
#pragma unroll
        for (int v = 0; v < 8; ++v) {
            const int mrow = v + 8 * hf;
            const float ig = sigf(acc[0][v]);
            const float fg = sigf(acc[1][v]);
            const float gg = tanhf(acc[2][v]);
            const float og = sigf(acc[3][v]);
            const float cn = fg * cacc[v] + ig * gg;
            const float hn = og * tanhf(cn);
            const bool mk = t < lenv[v];
            cacc[v] = mk ? cn : cacc[v];
            hreg[v] = mk ? hn : hreg[v];
            hbuf[nxt][mrow * H_DIM + u0 + n] = (__bf16)hreg[v];
            ys[(size_t)(t * 16 + mrow) * H_DIM + u0 + n] = (__bf16)(mk ? hn : 0.0f);
        }
        __syncthreads();
    }
}

// ------------------------------- launcher ----------------------------------

extern "C" void kernel_launch(void* const* d_in, const int* in_sizes, int n_in,
                              void* d_out, int out_size, void* d_ws, size_t ws_size,
                              hipStream_t stream) {
    const float* x       = (const float*)d_in[0];
    const int*   lengths = (const int*)d_in[1];
    const float* W1      = (const float*)d_in[2];
    const float* b1      = (const float*)d_in[3];
    const float* Wih     = (const float*)d_in[4];
    const float* Whh     = (const float*)d_in[5];
    const float* bih     = (const float*)d_in[6];
    const float* bhh     = (const float*)d_in[7];
    const float* W2      = (const float*)d_in[8];
    const float* b2      = (const float*)d_in[9];

    char* ws = (char*)d_ws;
    size_t off = 0;
    auto alloc = [&](size_t bytes) -> void* {
        void* p = ws + off;
        off = (off + bytes + 255) & ~(size_t)255;
        return p;
    };
    __bf16* xb    = (__bf16*)alloc((size_t)M_DIM * KP_DIM * 2);
    __bf16* h0    = (__bf16*)alloc((size_t)M_DIM * H_DIM * 2);
    __bf16* ys0   = (__bf16*)alloc((size_t)M_DIM * H_DIM * 2);
    __bf16* ys1   = (__bf16*)alloc((size_t)M_DIM * H_DIM * 2);
    __bf16* gibuf = (__bf16*)alloc((size_t)M_DIM * G4_DIM * 2);
    __bf16* W1b   = (__bf16*)alloc((size_t)H_DIM * KP_DIM * 2);
    __bf16* Wihb  = (__bf16*)alloc((size_t)2 * G4_DIM * H_DIM * 2);
    __bf16* Whhb  = (__bf16*)alloc((size_t)2 * G4_DIM * H_DIM * 2);
    __bf16* W2b   = (__bf16*)alloc((size_t)NP_DIM * H_DIM * 2);
    float*  biasA = (float*)alloc((size_t)2 * G4_DIM * 4);
    float*  bias2 = (float*)alloc((size_t)NP_DIM * 4);

    const int TPB = 256;
    // --- weight/input conversion ---
    k_pad_x<<<(M_DIM * KP_DIM + TPB - 1) / TPB, TPB, 0, stream>>>(x, xb);
    k_pad2d<<<(H_DIM * KP_DIM + TPB - 1) / TPB, TPB, 0, stream>>>(
        W1, W1b, IN_DIM, KP_DIM, H_DIM, H_DIM * KP_DIM);
    k_conv<<<(2 * G4_DIM * H_DIM + TPB - 1) / TPB, TPB, 0, stream>>>(
        Wih, Wihb, 2 * G4_DIM * H_DIM);
    k_conv<<<(2 * G4_DIM * H_DIM + TPB - 1) / TPB, TPB, 0, stream>>>(
        Whh, Whhb, 2 * G4_DIM * H_DIM);
    k_pad2d<<<(NP_DIM * H_DIM + TPB - 1) / TPB, TPB, 0, stream>>>(
        W2, W2b, H_DIM, H_DIM, OUT_DIM, NP_DIM * H_DIM);
    k_addvec<<<(2 * G4_DIM + TPB - 1) / TPB, TPB, 0, stream>>>(
        bih, bhh, biasA, 2 * G4_DIM);
    k_padbias<<<(NP_DIM + TPB - 1) / TPB, TPB, 0, stream>>>(b2, bias2, OUT_DIM, NP_DIM);

    // --- input projection: h0 = relu(xb @ W1b^T + b1) ---
    {
        dim3 grid(M_DIM / 16, 1);
        wmma_gemm<KP_DIM / 32, 0, 1><<<grid, 256, 0, stream>>>(
            xb, W1b, b1, (void*)h0, nullptr, H_DIM, H_DIM, 0);
    }

    // --- two LSTM layers ---
    const __bf16* layer_in = h0;
    __bf16* layer_out[2] = {ys0, ys1};
    for (int l = 0; l < 2; ++l) {
        dim3 grid(M_DIM / 16, G4_DIM / 512);
        wmma_gemm<H_DIM / 32, 1, 0><<<grid, 256, 0, stream>>>(
            layer_in, Wihb + (size_t)l * G4_DIM * H_DIM, biasA + (size_t)l * G4_DIM,
            (void*)gibuf, nullptr, G4_DIM, 0, 0);
        lstm_recurrent<<<1, 512, 0, stream>>>(
            gibuf, Whhb + (size_t)l * G4_DIM * H_DIM, lengths, layer_out[l]);
        layer_in = layer_out[l];
    }

    // --- output projection: y = mask * (ys1 @ W2b^T + b2), f32 ---
    {
        dim3 grid(M_DIM / 16, 1);
        wmma_gemm<H_DIM / 32, 2, 0><<<grid, 256, 0, stream>>>(
            ys1, W2b, bias2, d_out, lengths, NP_DIM, OUT_DIM, OUT_DIM);
    }
}
